// GCN_6914897347186
// MI455X (gfx1250) — matrix operand order repaired and checked by reading, not compiled, not run on hardware
//
#include <hip/hip_runtime.h>
#include <hip/hip_bf16.h>

#define NFEAT 512
#define NHID  128
#define NOUT  64

typedef __attribute__((ext_vector_type(16))) __bf16 v16bf;
typedef __attribute__((ext_vector_type(8)))  float  v8f;

// ---------------------------------------------------------------------------
// One-time: convert row-major f32 weight [K x Ncols] into bf16 laid out in
// exact per-lane WMMA B-fragment order:
//   out[((c*T + t)*32 + lane)*16 + idx] =
//       bf16( W[(t*32 + (idx>>3)*16 + (lane>>4)*8 + (idx&7)) * Ncols + c*16 + (lane&15)] )
// so the GEMM hot loop fetches a whole v16bf fragment with one vector load.
// ---------------------------------------------------------------------------
__global__ __launch_bounds__(256) void gcn_swizzle_w(const float* __restrict__ W,
                                                     __bf16* __restrict__ out,
                                                     int T, int Ncols, int total) {
    const int i = blockIdx.x * 256 + threadIdx.x;
    if (i >= total) return;
    const int idx  = i & 15;
    const int lane = (i >> 4) & 31;
    const int rest = i >> 9;
    const int t    = rest % T;
    const int c    = rest / T;
    const int k    = t * 32 + (idx >> 3) * 16 + (lane >> 4) * 8 + (idx & 7);
    const int col  = c * 16 + (lane & 15);
    out[i] = (__bf16)W[(size_t)k * Ncols + col];
}

// ---------------------------------------------------------------------------
// GEMM1: H1[N,128] = X[N,512] @ W1[512,128] (bf16 WMMA, f32 accumulate)
// 1 block = one 16-row tile. 8 waves; wave w owns columns [16w,16w+16).
// ---------------------------------------------------------------------------
__global__ __launch_bounds__(256) void gcn_gemm1(const float* __restrict__ X,
                                                 const __bf16* __restrict__ W1s,
                                                 float* __restrict__ H1) {
    __shared__ __bf16 As[16 * NFEAT];            // 16 KB bf16 A panel
    const int m0  = blockIdx.x * 16;
    const int tid = threadIdx.x;

    // panel is a contiguous 16*512-float region: float4 coalesced stage
    const float4* Xv = (const float4*)(X + (size_t)m0 * NFEAT);
#pragma unroll
    for (int it = 0; it < (16 * NFEAT / 4) / 256; ++it) {
        const int i = tid + it * 256;
        const float4 v = Xv[i];
        __bf16* p = &As[i * 4];
        p[0] = (__bf16)v.x; p[1] = (__bf16)v.y;
        p[2] = (__bf16)v.z; p[3] = (__bf16)v.w;
    }
    __syncthreads();

    const int ct   = tid >> 5;             // wave = col tile
    const int lane = tid & 31;
    const int g    = lane >> 4;
    const int mn   = lane & 15;
    const int n0   = ct * 16;
    const int T    = NFEAT / 32;           // 16 k-tiles

    v8f acc = {0.f, 0.f, 0.f, 0.f, 0.f, 0.f, 0.f, 0.f};

    for (int t = 0; t < T; ++t) {
        // A fragment: two contiguous 8-element bf16 runs -> 2x ds_load_b128
        v16bf a;
        const __bf16* ap = &As[mn * NFEAT + t * 32 + g * 8];
#pragma unroll
        for (int j = 0; j < 8; ++j) { a[j] = ap[j]; a[8 + j] = ap[16 + j]; }
        // B fragment: one 32-byte vector load from pre-swizzled weights
        const v16bf b = *(const v16bf*)(W1s + ((size_t)(ct * T + t) * 32 + lane) * 16);
        acc = __builtin_amdgcn_wmma_f32_16x16x32_bf16(false, a, false, b,
                                                      (short)0, acc, false, false);
    }

#pragma unroll
    for (int r = 0; r < 8; ++r)
        H1[(size_t)(m0 + r + 8 * g) * NHID + n0 + mn] = acc[r];
}

// ---------------------------------------------------------------------------
// GEMM2: H2[N,64] = relu(AGG1[N,128]) @ W2[128,64]; ReLU fused into staging.
// ---------------------------------------------------------------------------
__global__ __launch_bounds__(128) void gcn_gemm2(const float* __restrict__ AGG1,
                                                 const __bf16* __restrict__ W2s,
                                                 float* __restrict__ H2) {
    __shared__ __bf16 As[16 * NHID];             // 4 KB
    const int m0  = blockIdx.x * 16;
    const int tid = threadIdx.x;

    const float4* Av = (const float4*)(AGG1 + (size_t)m0 * NHID);
#pragma unroll
    for (int it = 0; it < (16 * NHID / 4) / 128; ++it) {
        const int i = tid + it * 128;
        const float4 v = Av[i];
        __bf16* p = &As[i * 4];
        p[0] = (__bf16)fmaxf(v.x, 0.f); p[1] = (__bf16)fmaxf(v.y, 0.f);
        p[2] = (__bf16)fmaxf(v.z, 0.f); p[3] = (__bf16)fmaxf(v.w, 0.f);
    }
    __syncthreads();

    const int ct   = tid >> 5;
    const int lane = tid & 31;
    const int g    = lane >> 4;
    const int mn   = lane & 15;
    const int n0   = ct * 16;
    const int T    = NHID / 32;            // 4 k-tiles

    v8f acc = {0.f, 0.f, 0.f, 0.f, 0.f, 0.f, 0.f, 0.f};

#pragma unroll
    for (int t = 0; t < T; ++t) {
        v16bf a;
        const __bf16* ap = &As[mn * NHID + t * 32 + g * 8];
#pragma unroll
        for (int j = 0; j < 8; ++j) { a[j] = ap[j]; a[8 + j] = ap[16 + j]; }
        const v16bf b = *(const v16bf*)(W2s + ((size_t)(ct * T + t) * 32 + lane) * 16);
        acc = __builtin_amdgcn_wmma_f32_16x16x32_bf16(false, a, false, b,
                                                      (short)0, acc, false, false);
    }

#pragma unroll
    for (int r = 0; r < 8; ++r)
        H2[(size_t)(m0 + r + 8 * g) * NOUT + n0 + mn] = acc[r];
}

// ---------------------------------------------------------------------------
// Seed aggregation buffer with bias (atomics accumulate on top of it).
// ---------------------------------------------------------------------------
__global__ __launch_bounds__(256) void gcn_init_bias(float* __restrict__ out,
                                                     const float* __restrict__ bias,
                                                     int total, int d) {
    const int i = blockIdx.x * 256 + threadIdx.x;
    if (i < total) out[i] = bias[i & (d - 1)];
}

// ---------------------------------------------------------------------------
// SpMM d=128: one wave per edge, float4 per lane, 4 f32 atomics (L2-resident).
// ---------------------------------------------------------------------------
__global__ __launch_bounds__(256) void gcn_spmm128(const int* __restrict__ esrc,
                                                   const int* __restrict__ edst,
                                                   const float* __restrict__ ew,
                                                   const float* __restrict__ H,
                                                   float* __restrict__ OUT,
                                                   int E) {
    const int t    = blockIdx.x * 256 + threadIdx.x;
    const int e    = t >> 5;
    const int lane = t & 31;
    if (e >= E) return;
    const int   s  = esrc[e];
    const int   d  = edst[e];
    const float wt = ew[e];
    const float4 v = ((const float4*)(H + (size_t)s * NHID))[lane];
    float* op = OUT + (size_t)d * NHID + lane * 4;
    atomicAdd(op + 0, wt * v.x);
    atomicAdd(op + 1, wt * v.y);
    atomicAdd(op + 2, wt * v.z);
    atomicAdd(op + 3, wt * v.w);
}

// ---------------------------------------------------------------------------
// SpMM d=64: one wave per edge, float2 per lane, 2 f32 atomics.
// ---------------------------------------------------------------------------
__global__ __launch_bounds__(256) void gcn_spmm64(const int* __restrict__ esrc,
                                                  const int* __restrict__ edst,
                                                  const float* __restrict__ ew,
                                                  const float* __restrict__ H,
                                                  float* __restrict__ OUT,
                                                  int E) {
    const int t    = blockIdx.x * 256 + threadIdx.x;
    const int e    = t >> 5;
    const int lane = t & 31;
    if (e >= E) return;
    const int   s  = esrc[e];
    const int   d  = edst[e];
    const float wt = ew[e];
    const float2 v = ((const float2*)(H + (size_t)s * NOUT))[lane];
    float* op = OUT + (size_t)d * NOUT + lane * 2;
    atomicAdd(op + 0, wt * v.x);
    atomicAdd(op + 1, wt * v.y);
}

extern "C" void kernel_launch(void* const* d_in, const int* in_sizes, int n_in,
                              void* d_out, int out_size, void* d_ws, size_t ws_size,
                              hipStream_t stream) {
    const float* x    = (const float*)d_in[0];
    const int*   esrc = (const int*)d_in[1];
    const int*   edst = (const int*)d_in[2];
    const float* ew   = (const float*)d_in[3];
    const float* W1   = (const float*)d_in[4];
    const float* b1   = (const float*)d_in[5];
    const float* W2   = (const float*)d_in[6];
    const float* b2   = (const float*)d_in[7];
    float* out = (float*)d_out;

    const int N = in_sizes[0] / NFEAT;   // 50000 (divisible by 16)
    const int E = in_sizes[1];           // 800000

    char* ws = (char*)d_ws;
    const size_t offH1  = 0;
    const size_t offAgg = (size_t)N * NHID * sizeof(float);            // 25.6 MB
    const size_t offH2  = 2 * (size_t)N * NHID * sizeof(float);        // 51.2 MB
    const size_t offW1s = offH2 + (size_t)N * NOUT * sizeof(float);    // 64.0 MB (32B aligned)
    const size_t offW2s = offW1s + (size_t)NFEAT * NHID * sizeof(__bf16);

    float*  h1   = (float*)(ws + offH1);
    float*  agg1 = (float*)(ws + offAgg);
    float*  h2   = (float*)(ws + offH2);
    __bf16* W1s  = (__bf16*)(ws + offW1s);
    __bf16* W2s  = (__bf16*)(ws + offW2s);

    // one-time weight convert + swizzle into WMMA B-fragment order
    gcn_swizzle_w<<<(NFEAT * NHID + 255) / 256, 256, 0, stream>>>(
        W1, W1s, NFEAT / 32, NHID, NFEAT * NHID);
    gcn_swizzle_w<<<(NHID * NOUT + 255) / 256, 256, 0, stream>>>(
        W2, W2s, NHID / 32, NOUT, NHID * NOUT);

    // layer 1
    gcn_gemm1<<<N / 16, 256, 0, stream>>>(x, W1s, h1);
    gcn_init_bias<<<(N * NHID + 255) / 256, 256, 0, stream>>>(agg1, b1, N * NHID, NHID);
    gcn_spmm128<<<(E * 32 + 255) / 256, 256, 0, stream>>>(esrc, edst, ew, h1, agg1, E);

    // layer 2 (ReLU fused into gemm2 staging)
    gcn_gemm2<<<N / 16, 128, 0, stream>>>(agg1, W2s, h2);
    gcn_init_bias<<<(N * NOUT + 255) / 256, 256, 0, stream>>>(out, b2, N * NOUT, NOUT);
    gcn_spmm64<<<(E * 32 + 255) / 256, 256, 0, stream>>>(esrc, edst, ew, h2, out, E);
}
